// QuaternionLinear_76922864271510
// MI455X (gfx1250) — matrix operand order repaired and checked by reading, not compiled, not run on hardware
//
#include <hip/hip_runtime.h>
#include <hip/hip_bf16.h>

typedef __attribute__((ext_vector_type(16))) __bf16 v16bf;
typedef __attribute__((ext_vector_type(2)))  __bf16 bf16x2;
typedef __attribute__((ext_vector_type(8)))  float  v8f;
typedef __attribute__((ext_vector_type(8)))  int    v8i;
typedef __attribute__((ext_vector_type(4)))  int    i4;
typedef __attribute__((ext_vector_type(2)))  int    i2;
typedef __attribute__((ext_vector_type(4)))  unsigned u32x4;
typedef __attribute__((ext_vector_type(8)))  int    i32x8;
typedef __attribute__((ext_vector_type(4)))  int    i32x4;

#define B_ROWS 8192
#define INQ    1024   // in_q
#define OUTQ   1024   // out_q
#define LDX    4096   // x row stride (4*in_q)
#define LDO    4096   // out row stride
#define KSTEP  32
#define MTILE  64     // rows per block
#define NTILE  128    // out_q cols per block
#define KPADU  20     // padded K extent in uints (32 bf16 -> 40 bf16 -> 20 uints)

static __device__ __forceinline__ unsigned pack2bf(float a, float b) {
    bf16x2 v;
    v[0] = (__bf16)a;
    v[1] = (__bf16)b;
    return __builtin_bit_cast(unsigned, v);
}

static __device__ __forceinline__ v8i ld_frag(const unsigned* p, int secondOffU) {
    i4 lo = *(const i4*)(p);
    i4 hi = *(const i4*)(p + secondOffU);
    return __builtin_shufflevector(lo, hi, 0, 1, 2, 3, 4, 5, 6, 7);
}

static __device__ __forceinline__ v16bf as_bf(v8i u) {
    return __builtin_bit_cast(v16bf, u);
}

#define WMMA_ACC(acc, A, Bf)                                                   \
    acc = __builtin_amdgcn_wmma_f32_16x16x32_bf16(false, as_bf(A), false,      \
                                                  as_bf(Bf), (short)0, acc,    \
                                                  false, false)

// ---------------------------------------------------------------------------
// TDM: issue one 2D tensor tile load (global bf16 -> LDS, padded rows).
// D# layout per CDNA5 ISA 8.3/8.4. tile rows are tile_d0 elements (2B each),
// tile_d1 rows, global row stride = stride0 elements. LDS gets 16B padding
// after every 64B (pad_interval code 3 = 16 DWORDs, pad_amount code 3 = 4
// DWORDs) -> 80B row stride == KPADU uints.
// ---------------------------------------------------------------------------
static __device__ __forceinline__ void tdm_load_2d(unsigned ldsOff,
                                                   const void* gaddr,
                                                   unsigned tile_d0,
                                                   unsigned tile_d1,
                                                   unsigned long long stride0) {
    unsigned long long ga = (unsigned long long)(uintptr_t)gaddr;
    u32x4 g0;
    g0[0] = 1u;                                           // count=1, user D#
    g0[1] = ldsOff;                                       // lds_addr (bytes)
    g0[2] = (unsigned)(ga & 0xFFFFFFFFu);                 // global_addr[31:0]
    g0[3] = (unsigned)((ga >> 32) & 0x1FFFFFFu) | (2u << 30); // [56:32] | type=2
    const unsigned w0 = (1u << 16)      // data_size = 2 bytes
                      | (1u << 20)      // pad_enable
                      | (3u << 22)      // pad_interval: 16 DWORDs (64B)
                      | (3u << 25);     // pad_amount:   4 DWORDs (16B)
    i32x8 g1;
    g1[0] = (int)w0;
    g1[1] = (int)((tile_d0 & 0xFFFFu) << 16);             // tensor_dim0[15:0]
    g1[2] = (int)(((tile_d0 >> 16) & 0xFFFFu) | ((tile_d1 & 0xFFFFu) << 16));
    g1[3] = (int)(((tile_d1 >> 16) & 0xFFFFu) | ((tile_d0 & 0xFFFFu) << 16));
    g1[4] = (int)(tile_d1 & 0xFFFFu);                     // tile_dim1, tile_dim2=0
    g1[5] = (int)(unsigned)(stride0 & 0xFFFFFFFFull);     // stride0[31:0]
    g1[6] = (int)(unsigned)((stride0 >> 32) & 0xFFFFull); // stride0[47:32]
    g1[7] = 0;                                            // stride1 = 0 (2D)
    i32x4 z4 = {0, 0, 0, 0};
    i32x8 z8 = {0, 0, 0, 0, 0, 0, 0, 0};
    // 6-arg form (clang-23 / therock-10.0 signature)
    __builtin_amdgcn_tensor_load_to_lds(g0, g1, z4, z4, z8, 0);
}

// ---------------------------------------------------------------------------
// Pass 1a: x fp32 -> bf16 (packed pairs), same layout
// ---------------------------------------------------------------------------
__global__ __launch_bounds__(256)
void convert_x_bf16(const float* __restrict__ x, unsigned* __restrict__ xb) {
    const size_t idx  = (size_t)blockIdx.x * blockDim.x + threadIdx.x;
    const size_t base = idx * 16;                 // 16 floats per thread
    const float4* p = (const float4*)(x + base);
    float4 v0 = p[0], v1 = p[1], v2 = p[2], v3 = p[3];
    i4 lo = { (int)pack2bf(v0.x, v0.y), (int)pack2bf(v0.z, v0.w),
              (int)pack2bf(v1.x, v1.y), (int)pack2bf(v1.z, v1.w) };
    i4 hi = { (int)pack2bf(v2.x, v2.y), (int)pack2bf(v2.z, v2.w),
              (int)pack2bf(v3.x, v3.y), (int)pack2bf(v3.z, v3.w) };
    i4* q = (i4*)(xb + base / 2);
    q[0] = lo;
    q[1] = hi;
}

// ---------------------------------------------------------------------------
// Pass 1b: W fp32 [k][n] -> bf16 transposed wt[w][n][k] (fragment-ready)
// ---------------------------------------------------------------------------
__global__ __launch_bounds__(256)
void transpose_w_bf16(const float* __restrict__ Wa, const float* __restrict__ Wb,
                      const float* __restrict__ Wc, const float* __restrict__ Wd,
                      unsigned* __restrict__ wt) {
    const float* Wp[4] = {Wa, Wb, Wc, Wd};
    const int w  = blockIdx.y;
    const int id = blockIdx.x * 256 + threadIdx.x;    // 0..65535
    const int k  = (id & 255) * 4;                    // K micro-tile
    const int n  = (id >> 8) * 4;                     // N micro-tile
    const float* W = Wp[w] + (size_t)k * OUTQ + n;
    float4 r0 = *(const float4*)(W + 0 * OUTQ);
    float4 r1 = *(const float4*)(W + 1 * OUTQ);
    float4 r2 = *(const float4*)(W + 2 * OUTQ);
    float4 r3 = *(const float4*)(W + 3 * OUTQ);
    float rr[4][4] = {{r0.x, r0.y, r0.z, r0.w}, {r1.x, r1.y, r1.z, r1.w},
                      {r2.x, r2.y, r2.z, r2.w}, {r3.x, r3.y, r3.z, r3.w}};
    #pragma unroll
    for (int nn = 0; nn < 4; ++nn) {
        i2 outv = { (int)pack2bf(rr[0][nn], rr[1][nn]),
                    (int)pack2bf(rr[2][nn], rr[3][nn]) };
        *(i2*)(wt + ((size_t)(w * OUTQ + n + nn) * INQ + k) / 2) = outv;
    }
}

// ---------------------------------------------------------------------------
// Pass 2: bf16 GEMM. TDM stages tiles into LDS (1 descriptor per wave per
// K-step), 64 WMMAs / wave / K-step.
// ---------------------------------------------------------------------------
__global__ __launch_bounds__(256)
void quat_gemm_bf16(const unsigned* __restrict__ xb,   // bf16 pairs [8192][2048]
                    const unsigned* __restrict__ wt,   // bf16 pairs [4][1024][512]
                    const float* __restrict__ bias,
                    float* __restrict__ out) {
    __shared__ unsigned sA[4 * MTILE * KPADU];   // 20 KB
    __shared__ unsigned sB[4 * NTILE * KPADU];   // 40 KB

    const int t       = threadIdx.x;
    const int rowBase = blockIdx.x * MTILE;
    const int colBase = blockIdx.y * NTILE;

    const int wid   = t >> 5;
    const int lane  = t & 31;
    const int waveM = wid & 1;
    const int waveN = wid >> 1;
    const int row16 = lane & 15;
    const int half  = lane >> 4;

    v8f acc[4][2][2];
    #pragma unroll
    for (int c = 0; c < 4; ++c)
        #pragma unroll
        for (int mt = 0; mt < 2; ++mt)
            #pragma unroll
            for (int nt = 0; nt < 2; ++nt)
                acc[c][mt][nt] = (v8f)0.0f;

    // --- per-wave TDM descriptor parameters (uniform within each wave) ---
    // waves 0..3: A tile for component wid; waves 4..7: B tile for weight wid-4
    const int  isB  = wid >> 2;
    const int  sel  = wid & 3;                         // component or weight idx
    const unsigned ldsA0 = (unsigned)(uintptr_t)(&sA[0]);
    const unsigned ldsB0 = (unsigned)(uintptr_t)(&sB[0]);
    const unsigned ldsOff =
        isB ? (ldsB0 + (unsigned)(sel * NTILE * KPADU * 4))
            : (ldsA0 + (unsigned)(sel * MTILE * KPADU * 4));
    const unsigned tileD1   = isB ? NTILE : MTILE;      // rows
    const unsigned long long strideEl = isB ? (unsigned long long)INQ
                                            : (unsigned long long)LDX;
    // element offset of tile start at k0=0 (in bf16 elements)
    const size_t gBaseEl =
        isB ? ((size_t)(sel * OUTQ + colBase) * INQ)
            : ((size_t)rowBase * LDX + (size_t)sel * INQ);
    const char* gBasePtr = isB ? (const char*)wt : (const char*)xb;

    const v8i sgn = (v8i)(int)0x80008000u;

    for (int k0 = 0; k0 < INQ; k0 += KSTEP) {
        // ---- stage via Tensor Data Mover: one 2D tile per wave ----
        tdm_load_2d(ldsOff, gBasePtr + (gBaseEl + (size_t)k0) * 2,
                    /*tile_d0=*/KSTEP, /*tile_d1=*/tileD1, strideEl);
        __builtin_amdgcn_s_wait_tensorcnt(0);
        asm volatile("" ::: "memory");   // TDM wrote LDS behind compiler's back
        __syncthreads();

        // ---- B fragments for both n-tiles, held in registers ----
        v8i uB[4][2];
        #pragma unroll
        for (int w = 0; w < 4; ++w)
            #pragma unroll
            for (int nt = 0; nt < 2; ++nt)
                uB[w][nt] = ld_frag(
                    sB + (w * NTILE + waveN * 32 + nt * 16 + row16) * KPADU + half * 4, 8);

        #pragma unroll
        for (int mt = 0; mt < 2; ++mt) {
            const int rIdx = waveM * 32 + mt * 16 + row16;
            v8i ua = ld_frag(sA + (0 * MTILE + rIdx) * KPADU + half * 4, 8);
            v8i ub = ld_frag(sA + (1 * MTILE + rIdx) * KPADU + half * 4, 8);
            v8i uc = ld_frag(sA + (2 * MTILE + rIdx) * KPADU + half * 4, 8);
            v8i ud = ld_frag(sA + (3 * MTILE + rIdx) * KPADU + half * 4, 8);
            v8i nb = ub ^ sgn;
            v8i nc = uc ^ sgn;
            v8i nd = ud ^ sgn;
            #pragma unroll
            for (int nt = 0; nt < 2; ++nt) {
                // out_a = a*Wa - b*Wb - c*Wc - d*Wd
                WMMA_ACC(acc[0][mt][nt], ua, uB[0][nt]);
                WMMA_ACC(acc[0][mt][nt], nb, uB[1][nt]);
                WMMA_ACC(acc[0][mt][nt], nc, uB[2][nt]);
                WMMA_ACC(acc[0][mt][nt], nd, uB[3][nt]);
                // out_b = a*Wb + b*Wa + c*Wd - d*Wc
                WMMA_ACC(acc[1][mt][nt], ua, uB[1][nt]);
                WMMA_ACC(acc[1][mt][nt], ub, uB[0][nt]);
                WMMA_ACC(acc[1][mt][nt], uc, uB[3][nt]);
                WMMA_ACC(acc[1][mt][nt], nd, uB[2][nt]);
                // out_c = a*Wc - b*Wd + c*Wa + d*Wb
                WMMA_ACC(acc[2][mt][nt], ua, uB[2][nt]);
                WMMA_ACC(acc[2][mt][nt], nb, uB[3][nt]);
                WMMA_ACC(acc[2][mt][nt], uc, uB[0][nt]);
                WMMA_ACC(acc[2][mt][nt], ud, uB[1][nt]);
                // out_d = a*Wd + b*Wc - c*Wb + d*Wa
                WMMA_ACC(acc[3][mt][nt], ua, uB[3][nt]);
                WMMA_ACC(acc[3][mt][nt], ub, uB[2][nt]);
                WMMA_ACC(acc[3][mt][nt], nc, uB[1][nt]);
                WMMA_ACC(acc[3][mt][nt], ud, uB[0][nt]);
            }
        }
        __syncthreads();
    }

    #pragma unroll
    for (int c = 0; c < 4; ++c) {
        #pragma unroll
        for (int nt = 0; nt < 2; ++nt) {
            const int col = c * OUTQ + colBase + waveN * 32 + nt * 16 + row16;
            const float bv = bias[col];
            #pragma unroll
            for (int mt = 0; mt < 2; ++mt) {
                const int rbase = rowBase + waveM * 32 + mt * 16 + half * 8;
                float* po = out + (size_t)rbase * LDO + col;
                #pragma unroll
                for (int r = 0; r < 8; ++r)
                    po[(size_t)r * LDO] = acc[c][mt][nt][r] + bv;
            }
        }
    }
}

// ---------------------------------------------------------------------------
// Fallback: fused fp32->bf16 conversion in the GEMM (used if ws too small)
// ---------------------------------------------------------------------------
__global__ __launch_bounds__(256)
void quat_linear_wmma(const float* __restrict__ x,
                      const float* __restrict__ Wa,
                      const float* __restrict__ Wb,
                      const float* __restrict__ Wc,
                      const float* __restrict__ Wd,
                      const float* __restrict__ bias,
                      float* __restrict__ out) {
    __shared__ unsigned sA[4 * MTILE * KPADU];
    __shared__ unsigned sB[4 * NTILE * KPADU];

    const int t       = threadIdx.x;
    const int rowBase = blockIdx.x * MTILE;
    const int colBase = blockIdx.y * NTILE;

    const int wid   = t >> 5;
    const int lane  = t & 31;
    const int waveM = wid & 1;
    const int waveN = wid >> 1;
    const int row16 = lane & 15;
    const int half  = lane >> 4;

    const float* Wp[4] = {Wa, Wb, Wc, Wd};

    v8f acc[4][2][2];
    #pragma unroll
    for (int c = 0; c < 4; ++c)
        #pragma unroll
        for (int mt = 0; mt < 2; ++mt)
            #pragma unroll
            for (int nt = 0; nt < 2; ++nt)
                acc[c][mt][nt] = (v8f)0.0f;

    const int aComp = t >> 6;
    const int aRow  = t & 63;
    const int kblk  = (t >> 5) & 7;
    const int nblk  = t & 31;

    const v8i sgn = (v8i)(int)0x80008000u;

    for (int k0 = 0; k0 < INQ; k0 += KSTEP) {
        {
            const float* gx = x + (size_t)(rowBase + aRow) * LDX + aComp * INQ + k0;
            unsigned* la = sA + (aComp * MTILE + aRow) * KPADU;
            #pragma unroll
            for (int q = 0; q < 8; ++q) {
                float4 v = ((const float4*)gx)[q];
                la[q * 2 + 0] = pack2bf(v.x, v.y);
                la[q * 2 + 1] = pack2bf(v.z, v.w);
            }
        }
        #pragma unroll
        for (int w = 0; w < 4; ++w) {
            const float* gw = Wp[w] + (size_t)(k0 + kblk * 4) * OUTQ + colBase + nblk * 4;
            float4 r0 = *(const float4*)(gw + 0 * OUTQ);
            float4 r1 = *(const float4*)(gw + 1 * OUTQ);
            float4 r2 = *(const float4*)(gw + 2 * OUTQ);
            float4 r3 = *(const float4*)(gw + 3 * OUTQ);
            unsigned* lb = sB + (w * NTILE + nblk * 4) * KPADU + kblk * 2;
            lb[0 * KPADU + 0] = pack2bf(r0.x, r1.x);
            lb[0 * KPADU + 1] = pack2bf(r2.x, r3.x);
            lb[1 * KPADU + 0] = pack2bf(r0.y, r1.y);
            lb[1 * KPADU + 1] = pack2bf(r2.y, r3.y);
            lb[2 * KPADU + 0] = pack2bf(r0.z, r1.z);
            lb[2 * KPADU + 1] = pack2bf(r2.z, r3.z);
            lb[3 * KPADU + 0] = pack2bf(r0.w, r1.w);
            lb[3 * KPADU + 1] = pack2bf(r2.w, r3.w);
        }
        __syncthreads();

        v8i uB[4][2];
        #pragma unroll
        for (int w = 0; w < 4; ++w)
            #pragma unroll
            for (int nt = 0; nt < 2; ++nt)
                uB[w][nt] = ld_frag(
                    sB + (w * NTILE + waveN * 32 + nt * 16 + row16) * KPADU + half * 4, 8);

        #pragma unroll
        for (int mt = 0; mt < 2; ++mt) {
            const int rIdx = waveM * 32 + mt * 16 + row16;
            v8i ua = ld_frag(sA + (0 * MTILE + rIdx) * KPADU + half * 4, 8);
            v8i ub = ld_frag(sA + (1 * MTILE + rIdx) * KPADU + half * 4, 8);
            v8i uc = ld_frag(sA + (2 * MTILE + rIdx) * KPADU + half * 4, 8);
            v8i ud = ld_frag(sA + (3 * MTILE + rIdx) * KPADU + half * 4, 8);
            v8i nb = ub ^ sgn;
            v8i nc = uc ^ sgn;
            v8i nd = ud ^ sgn;
            #pragma unroll
            for (int nt = 0; nt < 2; ++nt) {
                WMMA_ACC(acc[0][mt][nt], ua, uB[0][nt]);
                WMMA_ACC(acc[0][mt][nt], nb, uB[1][nt]);
                WMMA_ACC(acc[0][mt][nt], nc, uB[2][nt]);
                WMMA_ACC(acc[0][mt][nt], nd, uB[3][nt]);
                WMMA_ACC(acc[1][mt][nt], ua, uB[1][nt]);
                WMMA_ACC(acc[1][mt][nt], ub, uB[0][nt]);
                WMMA_ACC(acc[1][mt][nt], uc, uB[3][nt]);
                WMMA_ACC(acc[1][mt][nt], nd, uB[2][nt]);
                WMMA_ACC(acc[2][mt][nt], ua, uB[2][nt]);
                WMMA_ACC(acc[2][mt][nt], nb, uB[3][nt]);
                WMMA_ACC(acc[2][mt][nt], uc, uB[0][nt]);
                WMMA_ACC(acc[2][mt][nt], ud, uB[1][nt]);
                WMMA_ACC(acc[3][mt][nt], ua, uB[3][nt]);
                WMMA_ACC(acc[3][mt][nt], ub, uB[2][nt]);
                WMMA_ACC(acc[3][mt][nt], nc, uB[1][nt]);
                WMMA_ACC(acc[3][mt][nt], ud, uB[0][nt]);
            }
        }
        __syncthreads();
    }

    #pragma unroll
    for (int c = 0; c < 4; ++c) {
        #pragma unroll
        for (int nt = 0; nt < 2; ++nt) {
            const int col = c * OUTQ + colBase + waveN * 32 + nt * 16 + row16;
            const float bv = bias[col];
            #pragma unroll
            for (int mt = 0; mt < 2; ++mt) {
                const int rbase = rowBase + waveM * 32 + mt * 16 + half * 8;
                float* po = out + (size_t)rbase * LDO + col;
                #pragma unroll
                for (int r = 0; r < 8; ++r)
                    po[(size_t)r * LDO] = acc[c][mt][nt][r] + bv;
            }
        }
    }
}

extern "C" void kernel_launch(void* const* d_in, const int* in_sizes, int n_in,
                              void* d_out, int out_size, void* d_ws, size_t ws_size,
                              hipStream_t stream) {
    const float* x    = (const float*)d_in[0];
    const float* Wa   = (const float*)d_in[1];
    const float* Wb   = (const float*)d_in[2];
    const float* Wc   = (const float*)d_in[3];
    const float* Wd   = (const float*)d_in[4];
    const float* bias = (const float*)d_in[5];
    float* out = (float*)d_out;

    const size_t xbBytes = (size_t)B_ROWS * LDX * 2;          // 64 MB
    const size_t wtBytes = 4ull * OUTQ * INQ * 2;             // 8 MB
    dim3 gemmGrid(B_ROWS / MTILE, OUTQ / NTILE);              // 128 x 8

    if (ws_size >= xbBytes + wtBytes) {
        unsigned* xb = (unsigned*)d_ws;
        unsigned* wt = (unsigned*)((char*)d_ws + xbBytes);
        convert_x_bf16<<<dim3((B_ROWS * LDX) / (16 * 256)), dim3(256), 0, stream>>>(
            x, xb);
        transpose_w_bf16<<<dim3(256, 4), dim3(256), 0, stream>>>(Wa, Wb, Wc, Wd, wt);
        quat_gemm_bf16<<<gemmGrid, dim3(256), 0, stream>>>(xb, wt, bias, out);
    } else {
        quat_linear_wmma<<<gemmGrid, dim3(256), 0, stream>>>(
            x, Wa, Wb, Wc, Wd, bias, out);
    }
}
